// GNN_lstm_15522011808197
// MI455X (gfx1250) — matrix operand order repaired and checked by reading, not compiled, not run on hardware
//
#include <hip/hip_runtime.h>

// ---------------------------------------------------------------------------
// GNN + BiLSTM for MI455X (gfx1250, wave32, WMMA f32_16x16x32_f16)
// N=8192, T=20, M=6, D=9
// - LDS staging buffers padded to K-stride 32 and pre-zeroed: A-builds are
//   unconditional ds_load_b128 runs + v_cvt_pk_f16_f32.
// - Hardware-native transcendentals (v_exp_f32 + v_rcp_f32).
// - All recurring index math hoisted out of the sequential t-loops.
// ---------------------------------------------------------------------------

typedef __attribute__((ext_vector_type(16))) _Float16 v16h;
typedef __attribute__((ext_vector_type(8)))  float    v8f;

#define WMMA_F16(a, b, c) \
  __builtin_amdgcn_wmma_f32_16x16x32_f16(false, (a), false, (b), (short)0, (c), false, false)

__device__ __forceinline__ float frcp_(float x) { return __builtin_amdgcn_rcpf(x); }
__device__ __forceinline__ float fsig_(float x) { return frcp_(1.0f + __expf(-x)); }
// tanh(x) = 1 - 2/(exp(2x)+1); saturates correctly at +/-inf via rcp(inf)=0
__device__ __forceinline__ float ftanh_(float x) { return 1.0f - 2.0f * frcp_(1.0f + __expf(2.0f * x)); }

// ---- fragment builders (layouts per CDNA5 ISA 7.12.2, wave32) --------------
// A 16x32 f16 from padded (stride-32, zero-filled) LDS rows.
// lane l holds row (l&15); element j -> K = (j&7) + 16*(j>>3) + 8*(l>>4)
__device__ __forceinline__ v16h build_a32(const float* base, int rowBase, int lane) {
  int lr = lane & 15, hi = lane >> 4;
  const float* rp = base + (rowBase + lr) * 32 + (hi << 3);
  v16h a;
#pragma unroll
  for (int j = 0; j < 8; ++j) a[j] = (_Float16)rp[j];          // K = 8*hi + j
#pragma unroll
  for (int j = 0; j < 8; ++j) a[8 + j] = (_Float16)rp[16 + j]; // K = 16 + 8*hi + j
  return a;
}

// B 32x16 f16 for X @ W^T: B[k][n] = W[n][k]. Branch-free: clamped index + mask.
__device__ __forceinline__ v16h build_b_wt(const float* W, int outDim, int inDim, int co, int lane) {
  int n = (lane & 15) + co, hi = lane >> 4;
  int nc = (n < outDim) ? n : 0;
  float nm = (n < outDim) ? 1.0f : 0.0f;
  v16h b;
#pragma unroll
  for (int j = 0; j < 16; ++j) {
    int K = (j & 7) + (hi << 3) + ((j >> 3) << 4);
    int Kc = (K < inDim) ? K : 0;
    float km = (K < inDim) ? 1.0f : 0.0f;
    b[j] = (_Float16)(W[nc * inDim + Kc] * nm * km);
  }
  return b;
}

// B for [x|h] @ [Wih;Whh]^T stacked (K=0..8 -> Wih, K=9..17 -> Whh), outDim=36.
__device__ __forceinline__ v16h build_b_stack(const float* Wih, const float* Whh, int co, int lane) {
  int n = (lane & 15) + co, hi = lane >> 4;
  int nc = (n < 36) ? n : 0;
  float nm = (n < 36) ? 1.0f : 0.0f;
  v16h b;
#pragma unroll
  for (int j = 0; j < 16; ++j) {
    int K = (j & 7) + (hi << 3) + ((j >> 3) << 4);
    const float* src = (K < 9) ? Wih : Whh;
    int Kr = (K < 9) ? K : (K - 9);
    int Kc = (Kr < 9) ? Kr : 0;
    float km = (K < 18) ? 1.0f : 0.0f;
    b[j] = (_Float16)(src[nc * 9 + Kc] * nm * km);
  }
  return b;
}

// C init with bias broadcast along rows (value depends only on N = (l&15)+co).
__device__ __forceinline__ v8f cbias1(const float* b, int outDim, int co, int lane) {
  int n = (lane & 15) + co;
  int nc = (n < outDim) ? n : 0;
  float v = b[nc] * ((n < outDim) ? 1.0f : 0.0f);
  v8f c;
#pragma unroll
  for (int r = 0; r < 8; ++r) c[r] = v;
  return c;
}
__device__ __forceinline__ v8f cbias2(const float* b1, const float* b2, int outDim, int co, int lane) {
  int n = (lane & 15) + co;
  int nc = (n < outDim) ? n : 0;
  float m = (n < outDim) ? 1.0f : 0.0f;
  float v = (b1[nc] + b2[nc]) * m;
  v8f c;
#pragma unroll
  for (int r = 0; r < 8; ++r) c[r] = v;
  return c;
}

// D 16x16 f32 -> LDS: element r -> (M = r + 8*(l>>4), N = (l&15))
__device__ __forceinline__ void store_d_lds(float* base, int stride, int rowBase, v8f d, int co, int maxCol, int lane) {
  int n = (lane & 15), hi = lane >> 4;
  if (co + n < maxCol) {   // folds to unconditional when co+15 < maxCol
#pragma unroll
    for (int r = 0; r < 8; ++r)
      base[(rowBase + r + (hi << 3)) * stride + co + n] = d[r];
  }
}

// ---------------------------------------------------------------------------
// Kernel A: node features -> 2 rounds of attention-GRU -> hidden (ws) + pred_label0
// One wave = 8 cells (= 48 rows = 3 WMMA row tiles). Block = 64 threads = 2 waves.
// ---------------------------------------------------------------------------
__global__ __launch_bounds__(64) void gnn_kernel(
    const float* __restrict__ nodes_feature, const float* __restrict__ pos,
    const float* __restrict__ attmat, const int* __restrict__ num_rec,
    const float* __restrict__ msg_W, const float* __restrict__ msg_b,
    const float* __restrict__ gru_Wih, const float* __restrict__ gru_Whh,
    const float* __restrict__ gru_bih, const float* __restrict__ gru_bhh,
    const float* __restrict__ ro1_W, const float* __restrict__ ro1_b,
    const float* __restrict__ ro2_W, const float* __restrict__ ro2_b,
    float* __restrict__ hidden_ws, float* __restrict__ out_p0)
{
  __shared__ float h_s[2][48][32];    // hidden, padded for A-builds
  __shared__ float m_s[2][48][32];    // message-mix / relu buffer, padded
  __shared__ float msg_s[2][48][12];  // raw msg (VALU-read only)
  __shared__ float gx_s[2][48][28];
  __shared__ float gh_s[2][48][28];
  __shared__ float att_s[2][8][36];
  __shared__ float vm_s[2][48];

  const int lane = threadIdx.x & 31;
  const int w    = threadIdx.x >> 5;
  const long cellBase = (long)(blockIdx.x * 2 + w) * 8;   // cell index = n*T + t

  // Weight fragments (branch-free global loads, once per wave)
  v16h msgB  = build_b_wt(msg_W, 9, 9, 0, lane);
  v16h WihB0 = build_b_wt(gru_Wih, 27, 9, 0, lane);
  v16h WihB1 = build_b_wt(gru_Wih, 27, 9, 16, lane);
  v16h WhhB0 = build_b_wt(gru_Whh, 27, 9, 0, lane);
  v16h WhhB1 = build_b_wt(gru_Whh, 27, 9, 16, lane);
  v16h ro1B  = build_b_wt(ro1_W, 9, 9, 0, lane);
  v16h ro2B  = build_b_wt(ro2_W, 7, 9, 0, lane);

  float (*h)[32]   = h_s[w];
  float (*mmv)[32] = m_s[w];
  float (*mg)[12]  = msg_s[w];
  float (*gx)[28]  = gx_s[w];
  float (*gh)[28]  = gh_s[w];
  float (*att)[36] = att_s[w];
  float* vm        = vm_s[w];
  float* hF   = &h[0][0];
  float* mmF  = &mmv[0][0];
  float* mgF  = &mg[0][0];
  float* attF = &att[0][0];

  // Zero the padded staging arrays (padding columns must stay 0 forever)
  for (int i = lane; i < 48 * 32; i += 32) { hF[i] = 0.0f; mmF[i] = 0.0f; }
  __syncthreads();

  // Load node_feat = concat(nodes_feature, pos) and the validity mask
  for (int i = lane; i < 48; i += 32) {
    int c = i / 6, node = i % 6;
    long gc = cellBase + c;
    vm[i] = (node < num_rec[gc]) ? 1.0f : 0.0f;
    long b3 = (gc * 6 + node) * 3;
    long b6 = (gc * 6 + node) * 6;
    h[i][0] = nodes_feature[b3 + 0];
    h[i][1] = nodes_feature[b3 + 1];
    h[i][2] = nodes_feature[b3 + 2];
#pragma unroll
    for (int k = 0; k < 6; ++k) h[i][3 + k] = pos[b6 + k];
  }
  __syncthreads();

  // att = attmat * valid over source index q
  for (int i = lane; i < 8 * 36; i += 32) {
    int c = i / 36, pq = i % 36, q = pq % 6;
    long gc = cellBase + c;
    att[c][pq] = attmat[gc * 36 + pq] * vm[c * 6 + q];
  }
  __syncthreads();

  // Precompute lane-fixed index maps for the einsum (i = lane + 32k, i < 432)
  int eMOff[14], eAttB[14], eMgB[14];
#pragma unroll
  for (int k = 0; k < 14; ++k) {
    int i = lane + (k << 5);
    int ii = (i < 432) ? i : 0;
    int row = ii / 9, col = ii - row * 9;
    int c = row / 6, node = row - c * 6;
    eMOff[k] = row * 32 + col;       // write into m_s (stride 32)
    eAttB[k] = c * 36 + node * 6;    // att row base
    eMgB[k]  = c * 72 + col;         // mg: (c*6+q)*12 + col = c*72 + q*12 + col
  }

  // 2 rounds of message passing GRU
  for (int round = 0; round < 2; ++round) {
    // msg = h @ msg_W^T + msg_b
    {
      v8f cb = cbias1(msg_b, 9, 0, lane);
#pragma unroll
      for (int tile = 0; tile < 3; ++tile) {
        v16h a = build_a32(hF, tile * 16, lane);
        v8f dd = WMMA_F16(a, msgB, cb);
        store_d_lds(mgF, 12, tile * 16, dd, 0, 9, lane);
      }
    }
    __syncthreads();

    // m = att @ msg (per-cell 6x6 @ 6x9) — VALU, too small for WMMA
#pragma unroll
    for (int k = 0; k < 14; ++k) {
      if (k < 13 || lane < 16) {
        float acc = 0.0f;
#pragma unroll
        for (int q = 0; q < 6; ++q) acc += attF[eAttB[k] + q] * mgF[eMgB[k] + q * 12];
        mmF[eMOff[k]] = acc;
      }
    }
    __syncthreads();

    // gx = m @ Wih^T + bih ; gh = h @ Whh^T + bhh  (27 cols -> 2 col tiles)
#pragma unroll
    for (int tile = 0; tile < 3; ++tile) {
      v16h am = build_a32(mmF, tile * 16, lane);
      v16h ah = build_a32(hF, tile * 16, lane);
      v8f dd;
      dd = WMMA_F16(am, WihB0, cbias1(gru_bih, 27, 0,  lane)); store_d_lds(&gx[0][0], 28, tile * 16, dd, 0,  27, lane);
      dd = WMMA_F16(am, WihB1, cbias1(gru_bih, 27, 16, lane)); store_d_lds(&gx[0][0], 28, tile * 16, dd, 16, 27, lane);
      dd = WMMA_F16(ah, WhhB0, cbias1(gru_bhh, 27, 0,  lane)); store_d_lds(&gh[0][0], 28, tile * 16, dd, 0,  27, lane);
      dd = WMMA_F16(ah, WhhB1, cbias1(gru_bhh, 27, 16, lane)); store_d_lds(&gh[0][0], 28, tile * 16, dd, 16, 27, lane);
    }
    __syncthreads();

    // GRU elementwise + mask
    for (int row = lane; row < 48; row += 32) {
      float vmask = vm[row];
#pragma unroll
      for (int dcol = 0; dcol < 9; ++dcol) {
        float hv = h[row][dcol];
        float rr = fsig_(gx[row][dcol]      + gh[row][dcol]);
        float zz = fsig_(gx[row][9 + dcol]  + gh[row][9 + dcol]);
        float nn = ftanh_(gx[row][18 + dcol] + rr * gh[row][18 + dcol]);
        h[row][dcol] = ((1.0f - zz) * nn + zz * hv) * vmask;
      }
    }
    __syncthreads();
  }

  // hidden -> workspace for the BiLSTM (layout [n*T+t][m][9])
  for (int i = lane; i < 48 * 9; i += 32) {
    int row = i / 9, dcol = i - (i / 9) * 9;
    hidden_ws[(cellBase * 6) * 9 + row / 6 * 54 + (row % 6) * 9 + dcol] = h[row][dcol];
  }

  // readout0: relu(h @ ro1^T + b1) @ ro2^T + b2, masked. Reuse m_s (padding intact).
#pragma unroll
  for (int tile = 0; tile < 3; ++tile) {
    v16h a = build_a32(hF, tile * 16, lane);
    v8f dd = WMMA_F16(a, ro1B, cbias1(ro1_b, 9, 0, lane));
#pragma unroll
    for (int r = 0; r < 8; ++r) dd[r] = fmaxf(dd[r], 0.0f);
    store_d_lds(mmF, 32, tile * 16, dd, 0, 9, lane);
  }
  __syncthreads();

  const long base0 = 8192L * 20 * 6 * 7;  // pred_label0 follows pred_label
#pragma unroll
  for (int tile = 0; tile < 3; ++tile) {
    v16h a = build_a32(mmF, tile * 16, lane);
    v8f p = WMMA_F16(a, ro2B, cbias1(ro2_b, 7, 0, lane));
    int col = lane & 15, hi = lane >> 4;
    if (col < 7) {
#pragma unroll
      for (int r = 0; r < 8; ++r) {
        int row = tile * 16 + r + (hi << 3);
        long gc = cellBase + row / 6;
        int node = row % 6;
        out_p0[base0 + (gc * 6 + node) * 7 + col] = p[r] * vm[row];
      }
    }
  }
}

// ---------------------------------------------------------------------------
// Kernel B: BiLSTM over T=20 on 49,152 rows + fused readout -> pred_label
// One wave = 16 rows. [x|h] kept in one padded [16][32] buffer so the gate
// matmul is one A-build + 3 WMMAs per step. All index math hoisted out of t.
// ---------------------------------------------------------------------------
__global__ __launch_bounds__(64) void lstm_kernel(
    const float* __restrict__ hidden_ws, const int* __restrict__ num_rec,
    const float* __restrict__ lf_Wih, const float* __restrict__ lf_Whh,
    const float* __restrict__ lf_bih, const float* __restrict__ lf_bhh,
    const float* __restrict__ lb_Wih, const float* __restrict__ lb_Whh,
    const float* __restrict__ lb_bih, const float* __restrict__ lb_bhh,
    const float* __restrict__ lr1_W, const float* __restrict__ lr1_b,
    const float* __restrict__ lr2_W, const float* __restrict__ lr2_b,
    float* __restrict__ out_p)
{
  __shared__ float xh_s[2][16][32];    // cols 0..8 = x (or hf), 9..17 = h, rest 0
  __shared__ float rel_s[2][16][32];   // relu buffer, padded
  __shared__ float g_s[2][16][36];
  __shared__ float hf_s[2][20][16][9];
  __shared__ float vm_s[2][16];

  const int lane = threadIdx.x & 31;
  const int w    = threadIdx.x >> 5;
  const long rowBase = (long)(blockIdx.x * 2 + w) * 16;   // row = n*6 + m

  v16h Bf0 = build_b_stack(lf_Wih, lf_Whh, 0,  lane);
  v16h Bf1 = build_b_stack(lf_Wih, lf_Whh, 16, lane);
  v16h Bf2 = build_b_stack(lf_Wih, lf_Whh, 32, lane);
  v16h Bb0 = build_b_stack(lb_Wih, lb_Whh, 0,  lane);
  v16h Bb1 = build_b_stack(lb_Wih, lb_Whh, 16, lane);
  v16h Bb2 = build_b_stack(lb_Wih, lb_Whh, 32, lane);
  v16h R1B = build_b_wt(lr1_W, 9, 18, 0, lane);
  v16h R2B = build_b_wt(lr2_W, 7, 9,  0, lane);

  float (*xh)[32]    = xh_s[w];
  float (*rel)[32]   = rel_s[w];
  float (*g)[36]     = g_s[w];
  float (*hf)[16][9] = hf_s[w];
  float* vm          = vm_s[w];
  float* xhF = &xh[0][0];
  float* hfF = &hf[0][0][0];

  // -------- lane-fixed index maps (hoisted out of the t-loops) --------
  // Staging transfers: i = lane + 32k over 16x9 = 144 elements (k==4: lanes<16)
  int  xOff[5], hOff[5];
  long gOff[5];
#pragma unroll
  for (int k = 0; k < 5; ++k) {
    int i = lane + (k << 5);
    int ii = (i < 144) ? i : 0;
    int row = ii / 9, col = ii - row * 9;
    long R = rowBase + row;
    long n = R / 6; int m = (int)(R - n * 6);
    xOff[k] = row * 32 + col;                 // LDS offset into xh (and matching hf col)
    hOff[k] = row * 9 + col;                  // LDS offset into hf[t]
    gOff[k] = ((n * 20) * 6 + m) * 9 + col;   // global elem offset at t=0 (+ t*54)
  }
  // Validity-mask loader: lanes 0..15, one row each
  long vmN; int vmM;
  {
    long R = rowBase + (lane & 15);
    vmN = R / 6; vmM = (int)(R - vmN * 6);
  }
  // Output store map: element r -> row = r + 8*(l>>4), col = l&15
  long oOff[8];
  {
    int col = lane & 15, hi = lane >> 4;
#pragma unroll
    for (int r = 0; r < 8; ++r) {
      int row = r + (hi << 3);
      long R = rowBase + row;
      long n = R / 6; int m = (int)(R - n * 6);
      oOff[r] = ((n * 20) * 6 + m) * 7 + col;  // global elem offset at t=0 (+ t*42)
    }
  }

  // Zero padded buffers (h part of xh = initial hidden state 0)
  for (int i = lane; i < 16 * 32; i += 32) { xhF[i] = 0.0f; (&rel[0][0])[i] = 0.0f; }
  __syncthreads();

  float cc[9];
#pragma unroll
  for (int dcol = 0; dcol < 9; ++dcol) cc[dcol] = 0.0f;

  // -------- forward pass --------
  for (int t = 0; t < 20; ++t) {
    const float* hp = hidden_ws + (long)t * 54;
#pragma unroll
    for (int k = 0; k < 5; ++k)
      if (k < 4 || lane < 16) xhF[xOff[k]] = hp[gOff[k]];
    __syncthreads();
    v16h a = build_a32(xhF, 0, lane);
    v8f d0 = WMMA_F16(a, Bf0, cbias2(lf_bih, lf_bhh, 36, 0,  lane));
    v8f d1 = WMMA_F16(a, Bf1, cbias2(lf_bih, lf_bhh, 36, 16, lane));
    v8f d2 = WMMA_F16(a, Bf2, cbias2(lf_bih, lf_bhh, 36, 32, lane));
    store_d_lds(&g[0][0], 36, 0, d0, 0,  36, lane);
    store_d_lds(&g[0][0], 36, 0, d1, 16, 36, lane);
    store_d_lds(&g[0][0], 36, 0, d2, 32, 36, lane);
    __syncthreads();
    if (lane < 16) {
      int row = lane;
#pragma unroll
      for (int dcol = 0; dcol < 9; ++dcol) {
        float ig = fsig_(g[row][dcol]);
        float fg = fsig_(g[row][9 + dcol]);
        float gg = ftanh_(g[row][18 + dcol]);
        float og = fsig_(g[row][27 + dcol]);
        cc[dcol] = fg * cc[dcol] + ig * gg;
        float hh = og * ftanh_(cc[dcol]);
        xh[row][9 + dcol] = hh;      // h lives in cols 9..17
        hf[t][row][dcol] = hh;
      }
    }
    __syncthreads();
  }

  // -------- backward pass + fused readout --------
#pragma unroll
  for (int dcol = 0; dcol < 9; ++dcol) cc[dcol] = 0.0f;
  if (lane < 16) {
#pragma unroll
    for (int dcol = 0; dcol < 9; ++dcol) xh[lane][9 + dcol] = 0.0f;  // hb = 0
  }
  __syncthreads();

  for (int t = 19; t >= 0; --t) {
    const float* hp = hidden_ws + (long)t * 54;
#pragma unroll
    for (int k = 0; k < 5; ++k)
      if (k < 4 || lane < 16) xhF[xOff[k]] = hp[gOff[k]];
    if (lane < 16) vm[lane] = (vmM < num_rec[vmN * 20 + t]) ? 1.0f : 0.0f;
    __syncthreads();
    v16h a = build_a32(xhF, 0, lane);
    v8f d0 = WMMA_F16(a, Bb0, cbias2(lb_bih, lb_bhh, 36, 0,  lane));
    v8f d1 = WMMA_F16(a, Bb1, cbias2(lb_bih, lb_bhh, 36, 16, lane));
    v8f d2 = WMMA_F16(a, Bb2, cbias2(lb_bih, lb_bhh, 36, 32, lane));
    store_d_lds(&g[0][0], 36, 0, d0, 0,  36, lane);
    store_d_lds(&g[0][0], 36, 0, d1, 16, 36, lane);
    store_d_lds(&g[0][0], 36, 0, d2, 32, 36, lane);
    __syncthreads();
    if (lane < 16) {
      int row = lane;
#pragma unroll
      for (int dcol = 0; dcol < 9; ++dcol) {
        float ig = fsig_(g[row][dcol]);
        float fg = fsig_(g[row][9 + dcol]);
        float gg = ftanh_(g[row][18 + dcol]);
        float og = fsig_(g[row][27 + dcol]);
        cc[dcol] = fg * cc[dcol] + ig * gg;
        xh[row][9 + dcol] = og * ftanh_(cc[dcol]);   // hb in cols 9..17
      }
    }
    // overwrite x with hf(t): readout input becomes [hf(t) | hb]
    const float* hft = hfF + t * 144;
#pragma unroll
    for (int k = 0; k < 5; ++k)
      if (k < 4 || lane < 16) xhF[xOff[k]] = hft[hOff[k]];
    __syncthreads();

    // readout: relu([hf(t)|hb] @ lr1^T + b) @ lr2^T + b, masked
    v16h ar = build_a32(xhF, 0, lane);
    v8f r1 = WMMA_F16(ar, R1B, cbias1(lr1_b, 9, 0, lane));
#pragma unroll
    for (int r = 0; r < 8; ++r) r1[r] = fmaxf(r1[r], 0.0f);
    store_d_lds(&rel[0][0], 32, 0, r1, 0, 9, lane);
    __syncthreads();
    v16h a2 = build_a32(&rel[0][0], 0, lane);
    v8f p = WMMA_F16(a2, R2B, cbias1(lr2_b, 7, 0, lane));
    int col = lane & 15, hi = lane >> 4;
    if (col < 7) {
      float* op = out_p + (long)t * 42;
#pragma unroll
      for (int r = 0; r < 8; ++r)
        op[oOff[r]] = p[r] * vm[r + (hi << 3)];
    }
    __syncthreads();
  }
}

// ---------------------------------------------------------------------------
extern "C" void kernel_launch(void* const* d_in, const int* in_sizes, int n_in,
                              void* d_out, int out_size, void* d_ws, size_t ws_size,
                              hipStream_t stream) {
  (void)in_sizes; (void)n_in; (void)out_size; (void)ws_size;
  const float* nodes_feature = (const float*)d_in[0];
  const float* pos           = (const float*)d_in[1];
  const float* attmat        = (const float*)d_in[2];
  const float* msg_W         = (const float*)d_in[3];
  const float* msg_b         = (const float*)d_in[4];
  const float* gru_Wih       = (const float*)d_in[5];
  const float* gru_Whh       = (const float*)d_in[6];
  const float* gru_bih       = (const float*)d_in[7];
  const float* gru_bhh       = (const float*)d_in[8];
  const float* ro1_W         = (const float*)d_in[9];
  const float* ro1_b         = (const float*)d_in[10];
  const float* ro2_W         = (const float*)d_in[11];
  const float* ro2_b         = (const float*)d_in[12];
  const float* lf_Wih        = (const float*)d_in[13];
  const float* lf_Whh        = (const float*)d_in[14];
  const float* lf_bih        = (const float*)d_in[15];
  const float* lf_bhh        = (const float*)d_in[16];
  const float* lb_Wih        = (const float*)d_in[17];
  const float* lb_Whh        = (const float*)d_in[18];
  const float* lb_bih        = (const float*)d_in[19];
  const float* lb_bhh        = (const float*)d_in[20];
  const float* lr1_W         = (const float*)d_in[21];
  const float* lr1_b         = (const float*)d_in[22];
  const float* lr2_W         = (const float*)d_in[23];
  const float* lr2_b         = (const float*)d_in[24];
  const int*   num_rec       = (const int*)d_in[25];

  float* out       = (float*)d_out;
  float* hidden_ws = (float*)d_ws;   // N*T*M*9 floats = 35.4 MB

  // Kernel A: 163,840 cells / (2 waves * 8 cells) = 10,240 blocks of 64
  gnn_kernel<<<dim3(10240), dim3(64), 0, stream>>>(
      nodes_feature, pos, attmat, num_rec, msg_W, msg_b,
      gru_Wih, gru_Whh, gru_bih, gru_bhh,
      ro1_W, ro1_b, ro2_W, ro2_b, hidden_ws, out);

  // Kernel B: 49,152 rows / (2 waves * 16 rows) = 1,536 blocks of 64
  lstm_kernel<<<dim3(1536), dim3(64), 0, stream>>>(
      hidden_ws, num_rec,
      lf_Wih, lf_Whh, lf_bih, lf_bhh,
      lb_Wih, lb_Whh, lb_bih, lb_bhh,
      lr1_W, lr1_b, lr2_W, lr2_b, out);
}
